// SemanticTokenExtractor_74440373174617
// MI455X (gfx1250) — matrix operand order repaired and checked by reading, not compiled, not run on hardware
//
#include <hip/hip_runtime.h>

// ---------------------------------------------------------------------------
// SemanticTokenExtractor for MI455X (gfx1250, wave32, WMMA)
//   feats [16,4096,1024] f32, centroids [1000,1024] f32
//   tokens[B*N] int32  ++  embeddings[B*N,1024] f32  (concatenated in d_out)
//
// GEMM M=65536 x N=1000(K) x K=1024(D) as bf16x3 error-compensated
// v_wmma_f32_16x16x32_bf16 (xh*ch + xh*cl + xl*ch, f32 accumulate).
// Two centroid tiles per wave iteration reuse the A hi/lo fragments:
// 4 LDS + 8 global b128 per 6 WMMAs (2.0 loads/WMMA). hi/lo interleaved
// per row so every inner-loop access is base + immediate offset; LDS
// stride chosen so 16 fragment lanes hit distinct bank groups.
// ---------------------------------------------------------------------------

typedef __attribute__((ext_vector_type(16))) __bf16    v16bf;
typedef __attribute__((ext_vector_type(8)))  float     v8f;
typedef __attribute__((ext_vector_type(4)))  float     f4;
typedef __attribute__((ext_vector_type(4)))  unsigned  u32x4;

#define D_DIM      1024
#define K_REAL     1000
#define K_PAD      1024          // 64 tiles of 16 -> 32 tile-pairs
#define N_PAIRS    32
#define ROWS       64            // feature rows per workgroup
#define WAVES      4             // 4 waves * 16 rows = 64 rows
#define LDS_STRIDE 2056          // hi[0..1023] pad lo@1032[0..1023]; 1028 dw % 64 == 4
#define LDS_LO     1032          // lo offset inside a row (516 dw % 64 == 4)
#define B_STRIDE   2048          // workspace centroid row: hi[1024] ++ lo[1024]
#define B_TILE1    (16 * B_STRIDE)  // second tile of a pair: +64 KB immediate
#define M_TOTAL    65536         // B*N

union Frag16 { v16bf v; u32x4 q[2]; };

__device__ __forceinline__ unsigned short f2bf(float f) {
    unsigned u = __float_as_uint(f);
    unsigned r = (u + 0x7FFFu + ((u >> 16) & 1u)) >> 16;   // round-to-nearest-even
    return (unsigned short)r;
}
__device__ __forceinline__ float bf2f(unsigned short h) {
    return __uint_as_float(((unsigned)h) << 16);
}

// ---------------------------------------------------------------------------
// Prep: centroids f32 -> interleaved bf16 hi/lo rows [K_PAD][B_STRIDE] + ||c||^2
// ---------------------------------------------------------------------------
__global__ void stx_prep_centroids(const float* __restrict__ cent,
                                   unsigned short* __restrict__ cbuf,
                                   float* __restrict__ csq) {
    const int k   = blockIdx.x;        // 0..K_PAD-1
    const int tid = threadIdx.x;       // 256 threads, 4 elems each
    __shared__ float red[256];

    f4 x = (f4)(0.0f);
    if (k < K_REAL)
        x = ((const f4*)(cent + (size_t)k * D_DIM))[tid];

    unsigned short h[4], l[4];
    float p = 0.0f;
#pragma unroll
    for (int i = 0; i < 4; ++i) {
        h[i] = f2bf(x[i]);
        l[i] = f2bf(x[i] - bf2f(h[i]));
        p += x[i] * x[i];
    }
    unsigned* ph = (unsigned*)(cbuf + (size_t)k * B_STRIDE + tid * 4);
    unsigned* pl = (unsigned*)(cbuf + (size_t)k * B_STRIDE + D_DIM + tid * 4);
    ph[0] = (unsigned)h[0] | ((unsigned)h[1] << 16);
    ph[1] = (unsigned)h[2] | ((unsigned)h[3] << 16);
    pl[0] = (unsigned)l[0] | ((unsigned)l[1] << 16);
    pl[1] = (unsigned)l[2] | ((unsigned)l[3] << 16);

    red[tid] = p;
    __syncthreads();
    for (int s = 128; s > 0; s >>= 1) {
        if (tid < s) red[tid] += red[tid + s];
        __syncthreads();
    }
    if (tid == 0) csq[k] = (k < K_REAL) ? red[0] : 3.0e38f;  // pad never wins
}

// ---------------------------------------------------------------------------
// Main: stage 64 rows hi/lo in LDS once, sweep 32 centroid tile-pairs with
// 6x v_wmma_f32_16x16x32_bf16 per 32-K chunk, butterfly argmin, gather.
// ---------------------------------------------------------------------------
__global__ void __launch_bounds__(128)
stx_tokens_wmma(const float* __restrict__ feats,
                const float* __restrict__ cent,
                const unsigned short* __restrict__ cbuf,
                const float* __restrict__ csq,
                int* __restrict__ tok,
                float* __restrict__ emb) {
    extern __shared__ unsigned short smem[];

    const int tid  = threadIdx.x;
    const int lane = tid & 31;
    const int wave = tid >> 5;       // 0..3
    const int half = lane >> 4;      // 0 | 1
    const int l15  = lane & 15;

    // ---- stage 64 rows of feats as bf16 hi/lo into LDS (once) -------------
    const int rowg0 = blockIdx.x * ROWS;
    for (int i = tid; i < ROWS * (D_DIM / 4); i += blockDim.x) {
        const int r  = i >> 8;             // D/4 = 256 f4 per row
        const int k4 = i & 255;
        f4 x = __builtin_nontemporal_load(
            (const f4*)(feats + (size_t)(rowg0 + r) * D_DIM) + k4);
        unsigned short h[4], l[4];
#pragma unroll
        for (int e = 0; e < 4; ++e) {
            h[e] = f2bf(x[e]);
            l[e] = f2bf(x[e] - bf2f(h[e]));
        }
        unsigned* ph = (unsigned*)(smem + r * LDS_STRIDE + k4 * 4);
        unsigned* pl = (unsigned*)(smem + r * LDS_STRIDE + LDS_LO + k4 * 4);
        ph[0] = (unsigned)h[0] | ((unsigned)h[1] << 16);
        ph[1] = (unsigned)h[2] | ((unsigned)h[3] << 16);
        pl[0] = (unsigned)l[0] | ((unsigned)l[1] << 16);
        pl[1] = (unsigned)l[2] | ((unsigned)l[3] << 16);
    }
    __syncthreads();

    // ---- A-fragment base (16-bit A 16x32, contiguous K per lane half) ---------------
    // lane<16: row=l15, K {0..15}; lane>=16: row=l15, K {16..31}
    const unsigned short* aHi = smem + (wave * 16 + l15) * LDS_STRIDE + half * 16;

    float best[8];
    int   bestIdx[8];
#pragma unroll
    for (int r = 0; r < 8; ++r) { best[r] = 3.4e38f; bestIdx[r] = 0; }

    for (int jp = 0; jp < N_PAIRS; ++jp) {
        const int n0 = jp * 32 + l15;                          // tile 0 column
        const unsigned short* bHi = cbuf + (size_t)n0 * B_STRIDE + half * 16;
        const float csq0 = csq[n0];
        const float csq1 = csq[n0 + 16];

        v8f acc0 = (v8f)(0.0f);
        v8f acc1 = (v8f)(0.0f);
#pragma unroll 4
        for (int kc = 0; kc < 32; ++kc) {
            Frag16 ah, al, bh0, bl0, bh1, bl1;
            const u32x4* pah = (const u32x4*)(aHi + kc * 32);
            const u32x4* pal = (const u32x4*)(aHi + LDS_LO + kc * 32);
            ah.q[0] = pah[0]; ah.q[1] = pah[1];          // 16 contiguous K
            al.q[0] = pal[0]; al.q[1] = pal[1];
            const u32x4* ph0 = (const u32x4*)(bHi + kc * 32);
            const u32x4* pl0 = (const u32x4*)(bHi + D_DIM + kc * 32);
            const u32x4* ph1 = (const u32x4*)(bHi + B_TILE1 + kc * 32);
            const u32x4* pl1 = (const u32x4*)(bHi + B_TILE1 + D_DIM + kc * 32);
            bh0.q[0] = ph0[0]; bh0.q[1] = ph0[1];        // 16 contiguous K
            bl0.q[0] = pl0[0]; bl0.q[1] = pl0[1];
            bh1.q[0] = ph1[0]; bh1.q[1] = ph1[1];
            bl1.q[0] = pl1[0]; bl1.q[1] = pl1[1];

            acc0 = __builtin_amdgcn_wmma_f32_16x16x32_bf16(
                false, ah.v, false, bh0.v, (short)0, acc0, false, false);
            acc1 = __builtin_amdgcn_wmma_f32_16x16x32_bf16(
                false, ah.v, false, bh1.v, (short)0, acc1, false, false);
            acc0 = __builtin_amdgcn_wmma_f32_16x16x32_bf16(
                false, ah.v, false, bl0.v, (short)0, acc0, false, false);
            acc1 = __builtin_amdgcn_wmma_f32_16x16x32_bf16(
                false, ah.v, false, bl1.v, (short)0, acc1, false, false);
            acc0 = __builtin_amdgcn_wmma_f32_16x16x32_bf16(
                false, al.v, false, bh0.v, (short)0, acc0, false, false);
            acc1 = __builtin_amdgcn_wmma_f32_16x16x32_bf16(
                false, al.v, false, bh1.v, (short)0, acc1, false, false);
        }

        // dist = ||c||^2 - 2*score ; reduce over 16 columns of each tile.
        // C layout: VGPR r holds rows r (lanes 0-15) / r+8 (lanes 16-31),
        // column = lane&15 -> butterfly inside each 16-lane half.
#pragma unroll
        for (int t = 0; t < 2; ++t) {
            const float cs = t ? csq1 : csq0;
            const int   nb = n0 + t * 16;
#pragma unroll
            for (int r = 0; r < 8; ++r) {
                float dv = cs - 2.0f * (t ? acc1[r] : acc0[r]);
                int   di = nb;
#pragma unroll
                for (int m = 1; m < 16; m <<= 1) {
                    float od = __shfl_xor(dv, m, 32);
                    int   oi = __shfl_xor(di, m, 32);
                    if (od < dv || (od == dv && oi < di)) { dv = od; di = oi; }
                }
                if (dv < best[r] || (dv == best[r] && di < bestIdx[r])) {
                    best[r] = dv; bestIdx[r] = di;
                }
            }
        }
    }

    // ---- tokens ------------------------------------------------------------
    const int mbase = blockIdx.x * ROWS + wave * 16;
    if (l15 == 0) {
#pragma unroll
        for (int r = 0; r < 8; ++r)
            tok[mbase + half * 8 + r] = bestIdx[r];
    }

    // ---- embedding gather: wave copies its 16 rows (f32 centroids, L2-hot) -
#pragma unroll
    for (int rw = 0; rw < 16; ++rw) {
        const int t = __shfl(bestIdx[rw & 7], (rw < 8) ? 0 : 16, 32);
        const f4* src = (const f4*)(cent + (size_t)t * D_DIM);
        f4*       dst = (f4*)(emb + (size_t)(mbase + rw) * D_DIM);
        for (int c = lane; c < D_DIM / 4; c += 32)
            __builtin_nontemporal_store(src[c], dst + c);
    }
}

// ---------------------------------------------------------------------------
extern "C" void kernel_launch(void* const* d_in, const int* in_sizes, int n_in,
                              void* d_out, int out_size, void* d_ws, size_t ws_size,
                              hipStream_t stream) {
    const float* feats = (const float*)d_in[0];   // [16,4096,1024] f32
    const float* cent  = (const float*)d_in[1];   // [1000,1024]    f32

    char* ws = (char*)d_ws;
    unsigned short* cbuf = (unsigned short*)ws;                         // 4,194,304 B
    float*          csq  = (float*)(ws + (size_t)K_PAD * B_STRIDE * 2); //     4,096 B

    int*   tok = (int*)d_out;                 // first B*N int32 tokens
    float* emb = (float*)d_out + M_TOTAL;     // then B*N*D f32 embeddings

    stx_prep_centroids<<<K_PAD, 256, 0, stream>>>(cent, cbuf, csq);

    const size_t lds_bytes = (size_t)ROWS * LDS_STRIDE * sizeof(unsigned short); // 263,168 B
    stx_tokens_wmma<<<M_TOTAL / ROWS, WAVES * 32, lds_bytes, stream>>>(
        feats, cent, cbuf, csq, tok, emb);
}
